// ATPBlock_50672024158656
// MI455X (gfx1250) — matrix operand order repaired and checked by reading, not compiled, not run on hardware
//
#include <hip/hip_runtime.h>

typedef _Float16 half_t;
typedef _Float16 v16h __attribute__((ext_vector_type(16)));
typedef float v8f __attribute__((ext_vector_type(8)));

constexpr int kB = 2, kT = 2048, kD = 1024, kH = 16, kDH = 64;
constexpr int kM = kB * kT;              // 4096 token rows
constexpr float kScale = 0.125f;         // 1/sqrt(64)
constexpr float kEps = 1e-6f;
constexpr int kLdst = 40;                // GEMM LDS tile stride (halves): 16B-aligned rows

union FragA { v16h v; unsigned u[8]; };
union FragC { v8f v; float f[8]; };

__device__ __forceinline__ int a_koff(int p, int hi) {
  // 16-bit A-matrix 16x32 layout (contiguous): lanes 0-15 K={0..15}, lanes 16-31 K={16..31}
  return hi * 16 + p * 2;
}

__device__ __forceinline__ v8f zero8() {
  FragC c;
#pragma unroll
  for (int i = 0; i < 8; ++i) c.f[i] = 0.f;
  return c.v;
}

// ---- CDNA5 async global->LDS staging (ASYNCcnt path), with safe fallback ----
#if defined(__has_builtin)
#if __has_builtin(__builtin_amdgcn_global_load_async_to_lds_b128)
#define USE_ASYNC_LDS 1
#endif
#endif

#if defined(USE_ASYNC_LDS)
typedef int v4i_t __attribute__((vector_size(16)));
typedef __attribute__((address_space(1))) v4i_t global_v4i;
typedef __attribute__((address_space(3))) v4i_t lds_v4i;
__device__ __forceinline__ void cp_async16(const half_t* g, half_t* l) {
  global_v4i* gp = (global_v4i*)(unsigned long long)(uintptr_t)g;
  lds_v4i* lp = (lds_v4i*)(unsigned)(uintptr_t)l;
  __builtin_amdgcn_global_load_async_to_lds_b128(gp, lp, 0, 0);
}
__device__ __forceinline__ void async_wait0() {
#if __has_builtin(__builtin_amdgcn_s_wait_asynccnt)
  __builtin_amdgcn_s_wait_asynccnt(0);
#else
  asm volatile("s_wait_asynccnt 0" ::: "memory");
#endif
}
#else
__device__ __forceinline__ void cp_async16(const half_t* g, half_t* l) {
  *(uint4*)l = *(const uint4*)g;
}
__device__ __forceinline__ void async_wait0() {}
#endif

// ---------------------------------------------------------------- cvt f32->f16
__global__ __launch_bounds__(256) void cvt_kernel(const float* __restrict__ in,
                                                  half_t* __restrict__ out, int n) {
  int i = blockIdx.x * 256 + threadIdx.x;
  if (i < n) out[i] = (half_t)in[i];
}

// ---------------------------------------------------------------- rmsnorm -> f16
__global__ __launch_bounds__(256) void rmsnorm_kernel(const float* __restrict__ x,
                                                      const float* __restrict__ w,
                                                      half_t* __restrict__ out) {
  const int row = blockIdx.x;
  const int tid = threadIdx.x;
  const float* xr = x + (size_t)row * kD;
  float v[4];
  float s = 0.f;
#pragma unroll
  for (int i = 0; i < 4; ++i) {
    v[i] = xr[tid + i * 256];
    s += v[i] * v[i];
  }
  __shared__ float red[256];
  red[tid] = s;
  __syncthreads();
  for (int ofs = 128; ofs > 0; ofs >>= 1) {
    if (tid < ofs) red[tid] += red[tid + ofs];
    __syncthreads();
  }
  const float r = rsqrtf(red[0] / (float)kD + kEps);
#pragma unroll
  for (int i = 0; i < 4; ++i) {
    const int c = tid + i * 256;
    out[(size_t)row * kD + c] = (half_t)(v[i] * r * w[c]);
  }
}

// ---------------------------------------------------------------- generic WMMA GEMM
// C[M,N] = A[M,K] * W[N,K]^T ; 128x128 block tile, 8 waves of 32x64, BLK_K=32.
enum { EP_QKV = 0, EP_RESID = 1 };

__global__ __launch_bounds__(256) void gemm_kernel(
    const half_t* __restrict__ A, const half_t* __restrict__ Bw, int M, int N, int K,
    int mode, const float* __restrict__ resid, float* __restrict__ outf,
    half_t* __restrict__ q16, half_t* __restrict__ k16, half_t* __restrict__ v16,
    float* __restrict__ kc, float* __restrict__ vc) {
  __shared__ half_t As[128 * kLdst];
  __shared__ half_t Bs[128 * kLdst];
  const int tid = threadIdx.x;
  const int bm = blockIdx.y * 128;
  const int bn = blockIdx.x * 128;
  const int wid = tid >> 5, lane = tid & 31;
  const int wm = wid & 3, wn = wid >> 2;
  const int lm = lane & 15, hi = lane >> 4;

  v8f acc[2][4];
#pragma unroll
  for (int mt = 0; mt < 2; ++mt)
#pragma unroll
    for (int nt = 0; nt < 4; ++nt) acc[mt][nt] = zero8();

  for (int k0 = 0; k0 < K; k0 += 32) {
#pragma unroll
    for (int pass = 0; pass < 2; ++pass) {
      const int i = tid + pass * 256;  // 512 uint4 per tile
      const int row = i >> 2;
      const int col = (i & 3) * 8;
      cp_async16(A + (size_t)(bm + row) * K + k0 + col, As + row * kLdst + col);
      cp_async16(Bw + (size_t)(bn + row) * K + k0 + col, Bs + row * kLdst + col);
    }
    async_wait0();
    __syncthreads();

    FragA af[2], bf4[4];
#pragma unroll
    for (int mt = 0; mt < 2; ++mt) {
      const half_t* ap = As + (wm * 32 + mt * 16 + lm) * kLdst;
#pragma unroll
      for (int p = 0; p < 8; ++p) af[mt].u[p] = *(const unsigned*)(ap + a_koff(p, hi));
    }
#pragma unroll
    for (int nt = 0; nt < 4; ++nt) {
      const half_t* bp = Bs + (wn * 64 + nt * 16 + lm) * kLdst;
#pragma unroll
      for (int p = 0; p < 8; ++p) bf4[nt].u[p] = *(const unsigned*)(bp + hi * 16 + 2 * p);
    }
#pragma unroll
    for (int mt = 0; mt < 2; ++mt)
#pragma unroll
      for (int nt = 0; nt < 4; ++nt)
        acc[mt][nt] = __builtin_amdgcn_wmma_f32_16x16x32_f16(
            false, af[mt].v, false, bf4[nt].v, (short)0, acc[mt][nt], false, false);
    __syncthreads();
  }

#pragma unroll
  for (int mt = 0; mt < 2; ++mt)
#pragma unroll
    for (int nt = 0; nt < 4; ++nt) {
      FragC c;
      c.v = acc[mt][nt];
      const int ng = bn + wn * 64 + nt * 16 + lm;
#pragma unroll
      for (int r = 0; r < 8; ++r) {
        const int mg = bm + wm * 32 + mt * 16 + r + 8 * hi;
        if (mode == EP_RESID) {
          const size_t idx = (size_t)mg * N + ng;
          outf[idx] = c.f[r] + resid[idx];
        } else {  // EP_QKV scatter
          const int bb = mg >> 11;       // / T
          const int t = mg & 2047;
          const int which = ng >> 10;    // / D
          const int hd = ng & 1023;
          const int hh = hd >> 6;
          const int dh = hd & 63;
          const size_t hidx = (((size_t)(bb * kH + hh)) * kT + t) * kDH + dh;
          if (which == 0) {
            q16[hidx] = (half_t)c.f[r];
          } else if (which == 1) {
            k16[hidx] = (half_t)c.f[r];
            kc[hidx] = c.f[r];
          } else {
            v16[hidx] = (half_t)c.f[r];
            vc[hidx] = c.f[r];
          }
        }
      }
    }
}

// ---------------------------------------------------------------- fused gate/up GEMM
__global__ __launch_bounds__(256) void gateup_kernel(
    const half_t* __restrict__ A, const half_t* __restrict__ Wg,
    const half_t* __restrict__ Wu, half_t* __restrict__ gu, int N, int K) {
  __shared__ half_t As[128 * kLdst];
  __shared__ half_t Gs[128 * kLdst];
  __shared__ half_t Us[128 * kLdst];
  const int tid = threadIdx.x;
  const int bm = blockIdx.y * 128;
  const int bn = blockIdx.x * 128;
  const int wid = tid >> 5, lane = tid & 31;
  const int wm = wid & 3, wn = wid >> 2;
  const int lm = lane & 15, hi = lane >> 4;

  v8f ag[2][4], au[2][4];
#pragma unroll
  for (int mt = 0; mt < 2; ++mt)
#pragma unroll
    for (int nt = 0; nt < 4; ++nt) { ag[mt][nt] = zero8(); au[mt][nt] = zero8(); }

  for (int k0 = 0; k0 < K; k0 += 32) {
#pragma unroll
    for (int pass = 0; pass < 2; ++pass) {
      const int i = tid + pass * 256;
      const int row = i >> 2;
      const int col = (i & 3) * 8;
      cp_async16(A + (size_t)(bm + row) * K + k0 + col, As + row * kLdst + col);
      cp_async16(Wg + (size_t)(bn + row) * K + k0 + col, Gs + row * kLdst + col);
      cp_async16(Wu + (size_t)(bn + row) * K + k0 + col, Us + row * kLdst + col);
    }
    async_wait0();
    __syncthreads();

    FragA af[2];
#pragma unroll
    for (int mt = 0; mt < 2; ++mt) {
      const half_t* ap = As + (wm * 32 + mt * 16 + lm) * kLdst;
#pragma unroll
      for (int p = 0; p < 8; ++p) af[mt].u[p] = *(const unsigned*)(ap + a_koff(p, hi));
    }
#pragma unroll
    for (int nt = 0; nt < 4; ++nt) {
      FragA gf, uf;
      const half_t* gp = Gs + (wn * 64 + nt * 16 + lm) * kLdst;
      const half_t* up = Us + (wn * 64 + nt * 16 + lm) * kLdst;
#pragma unroll
      for (int p = 0; p < 8; ++p) {
        gf.u[p] = *(const unsigned*)(gp + hi * 16 + 2 * p);
        uf.u[p] = *(const unsigned*)(up + hi * 16 + 2 * p);
      }
#pragma unroll
      for (int mt = 0; mt < 2; ++mt) {
        ag[mt][nt] = __builtin_amdgcn_wmma_f32_16x16x32_f16(
            false, af[mt].v, false, gf.v, (short)0, ag[mt][nt], false, false);
        au[mt][nt] = __builtin_amdgcn_wmma_f32_16x16x32_f16(
            false, af[mt].v, false, uf.v, (short)0, au[mt][nt], false, false);
      }
    }
    __syncthreads();
  }

#pragma unroll
  for (int mt = 0; mt < 2; ++mt)
#pragma unroll
    for (int nt = 0; nt < 4; ++nt) {
      FragC cg, cu;
      cg.v = ag[mt][nt];
      cu.v = au[mt][nt];
      const int ng = bn + wn * 64 + nt * 16 + lm;
#pragma unroll
      for (int r = 0; r < 8; ++r) {
        const int mg = bm + wm * 32 + mt * 16 + r + 8 * hi;
        const float g = cg.f[r];
        const float s = g / (1.f + __expf(-g));  // silu
        gu[(size_t)mg * N + ng] = (half_t)(s * cu.f[r]);
      }
    }
}

// ---------------------------------------------------------------- flash attention
// grid = B*H*(T/64); 4 waves/block, each wave owns 16 query rows.
__global__ __launch_bounds__(128) void attn_kernel(const half_t* __restrict__ q16,
                                                   const half_t* __restrict__ k16,
                                                   const half_t* __restrict__ v16,
                                                   half_t* __restrict__ outh) {
  constexpr int NQB = kT / 64;
  const int qb = blockIdx.x % NQB;
  const int bh = blockIdx.x / NQB;
  const int b = bh / kH;
  const int h = bh % kH;

  const half_t* qp = q16 + (size_t)bh * kT * kDH;
  const half_t* kp = k16 + (size_t)bh * kT * kDH;
  const half_t* vp = v16 + (size_t)bh * kT * kDH;

  const int tid = threadIdx.x;
  const int wid = tid >> 5;
  const int lane = tid & 31;
  const int lm = lane & 15;
  const int hi = lane >> 4;
  const int qbase = qb * 64 + wid * 16;

  __shared__ half_t Ks[32 * 72];       // K tile, token-major [32][64] (pad 8, 16B rows)
  __shared__ half_t Vt[64 * 34];       // V tile transposed [dh][token] (pad 2)
  __shared__ half_t Ps[4 * 16 * 36];   // per-wave P bounce buffers
  half_t* ps = Ps + wid * (16 * 36);

  FragA qf[2];
  {
    const half_t* qrow = qp + (size_t)(qbase + lm) * kDH;
#pragma unroll
    for (int kq = 0; kq < 2; ++kq)
#pragma unroll
      for (int p = 0; p < 8; ++p)
        qf[kq].u[p] = *(const unsigned*)(qrow + kq * 32 + a_koff(p, hi));
  }

  float rmax[8], rsum[8], fac[8];
  v8f o[4];
#pragma unroll
  for (int r = 0; r < 8; ++r) { rmax[r] = -1e30f; rsum[r] = 0.f; }
#pragma unroll
  for (int d = 0; d < 4; ++d) o[d] = zero8();

  const int kblocks = qb * 2 + 2;
  for (int kb = 0; kb < kblocks; ++kb) {
    const int kt0 = kb * 32;
#pragma unroll
    for (int pass = 0; pass < 2; ++pass) {
      const int i = tid + pass * 128;  // 256 uint4 per tile
      const int row = i >> 3;
      const int col = (i & 7) * 8;
      cp_async16(kp + (size_t)(kt0 + row) * kDH + col, Ks + row * 72 + col);
      uint4 dv = *(const uint4*)(vp + (size_t)(kt0 + row) * kDH + col);
      const half_t* hv = (const half_t*)&dv;
#pragma unroll
      for (int j = 0; j < 8; ++j) Vt[(col + j) * 34 + row] = hv[j];
    }
    async_wait0();
    __syncthreads();

    if (kt0 <= qbase + 15) {
      // S = Q * K^T for 32 keys (two 16x16 tiles)
      FragC sf[2];
#pragma unroll
      for (int nt = 0; nt < 2; ++nt) {
        v8f s = zero8();
#pragma unroll
        for (int kq = 0; kq < 2; ++kq) {
          FragA bk;
          const half_t* kr = Ks + (nt * 16 + lm) * 72 + kq * 32;
#pragma unroll
          for (int p = 0; p < 8; ++p) bk.u[p] = *(const unsigned*)(kr + hi * 16 + 2 * p);
          s = __builtin_amdgcn_wmma_f32_16x16x32_f16(false, qf[kq].v, false, bk.v,
                                                     (short)0, s, false, false);
        }
        sf[nt].v = s;
      }
      // causal mask + scale + online softmax (row r lives in 16-lane half `hi`)
#pragma unroll
      for (int r = 0; r < 8; ++r) {
        const int mg = qbase + r + 8 * hi;
#pragma unroll
        for (int nt = 0; nt < 2; ++nt) {
          const int kt = kt0 + nt * 16 + lm;
          const float v = sf[nt].f[r] * kScale;
          sf[nt].f[r] = (kt > mg) ? -1e30f : v;
        }
        float mx = fmaxf(sf[0].f[r], sf[1].f[r]);
#pragma unroll
        for (int ofs = 1; ofs < 16; ofs <<= 1) mx = fmaxf(mx, __shfl_xor(mx, ofs, 32));
        const float nm = fmaxf(rmax[r], mx);
        const float p0 = __expf(sf[0].f[r] - nm);
        const float p1 = __expf(sf[1].f[r] - nm);
        sf[0].f[r] = p0;
        sf[1].f[r] = p1;
        float ls = p0 + p1;
#pragma unroll
        for (int ofs = 1; ofs < 16; ofs <<= 1) ls += __shfl_xor(ls, ofs, 32);
        const float f = __expf(rmax[r] - nm);
        rsum[r] = rsum[r] * f + ls;
        rmax[r] = nm;
        fac[r] = f;
      }
#pragma unroll
      for (int d = 0; d < 4; ++d) {
        FragC t;
        t.v = o[d];
#pragma unroll
        for (int r = 0; r < 8; ++r) t.f[r] *= fac[r];
        o[d] = t.v;
      }
      // P (C-layout) -> LDS -> A-layout fragment
#pragma unroll
      for (int nt = 0; nt < 2; ++nt)
#pragma unroll
        for (int r = 0; r < 8; ++r)
          ps[(r + 8 * hi) * 36 + nt * 16 + lm] = (half_t)sf[nt].f[r];
      asm volatile("s_wait_dscnt 0" ::: "memory");
      FragA pf;
#pragma unroll
      for (int p = 0; p < 8; ++p)
        pf.u[p] = *(const unsigned*)(ps + lm * 36 + a_koff(p, hi));
      // O += P * V
#pragma unroll
      for (int d = 0; d < 4; ++d) {
        FragA vf;
        const half_t* vr = Vt + (d * 16 + lm) * 34;
#pragma unroll
        for (int p = 0; p < 8; ++p) vf.u[p] = *(const unsigned*)(vr + hi * 16 + 2 * p);
        o[d] = __builtin_amdgcn_wmma_f32_16x16x32_f16(false, pf.v, false, vf.v,
                                                      (short)0, o[d], false, false);
      }
    }
    __syncthreads();
  }

#pragma unroll
  for (int d = 0; d < 4; ++d) {
    FragC t;
    t.v = o[d];
#pragma unroll
    for (int r = 0; r < 8; ++r) {
      const int trow = qbase + r + 8 * hi;
      const int dh = d * 16 + lm;
      outh[(size_t)(b * kT + trow) * kD + h * kDH + dh] = (half_t)(t.f[r] / rsum[r]);
    }
  }
}

// ---------------------------------------------------------------- launcher
extern "C" void kernel_launch(void* const* d_in, const int* in_sizes, int n_in,
                              void* d_out, int out_size, void* d_ws, size_t ws_size,
                              hipStream_t stream) {
  const float* x = (const float*)d_in[0];
  const float* wn1 = (const float*)d_in[1];
  const float* wqkv = (const float*)d_in[2];
  const float* wproj = (const float*)d_in[3];
  const float* wn2 = (const float*)d_in[4];
  const float* wgate = (const float*)d_in[5];
  const float* wup = (const float*)d_in[6];
  const float* wdown = (const float*)d_in[7];
  (void)in_sizes; (void)n_in; (void)out_size; (void)ws_size;

  char* ws = (char*)d_ws;
  size_t off = 0;
  auto take = [&](size_t bytes) {
    char* p = ws + off;
    off = (off + bytes + 255) & ~(size_t)255;
    return p;
  };

  half_t* wqkv16 = (half_t*)take((size_t)3 * kD * kD * 2);
  half_t* wproj16 = (half_t*)take((size_t)kD * kD * 2);
  half_t* wgate16 = (half_t*)take((size_t)4 * kD * kD * 2);
  half_t* wup16 = (half_t*)take((size_t)4 * kD * kD * 2);
  half_t* wdown16 = (half_t*)take((size_t)4 * kD * kD * 2);
  half_t* h16 = (half_t*)take((size_t)kM * kD * 2);       // reused for h2
  half_t* q16 = (half_t*)take((size_t)kM * kD * 2);
  half_t* k16 = (half_t*)take((size_t)kM * kD * 2);
  half_t* v16 = (half_t*)take((size_t)kM * kD * 2);
  half_t* attn16 = (half_t*)take((size_t)kM * kD * 2);
  half_t* gu16 = (half_t*)take((size_t)kM * 4 * kD * 2);
  float* x1 = (float*)take((size_t)kM * kD * 4);

  float* out_x = (float*)d_out;
  float* kcache = out_x + (size_t)kM * kD;
  float* vcache = kcache + (size_t)kM * kD;

  // 1) weights -> f16
  auto cvt = [&](const float* in, half_t* o, int n) {
    cvt_kernel<<<(n + 255) / 256, 256, 0, stream>>>(in, o, n);
  };
  cvt(wqkv, wqkv16, 3 * kD * kD);
  cvt(wproj, wproj16, kD * kD);
  cvt(wgate, wgate16, 4 * kD * kD);
  cvt(wup, wup16, 4 * kD * kD);
  cvt(wdown, wdown16, 4 * kD * kD);

  // 2) rmsnorm1
  rmsnorm_kernel<<<kM, 256, 0, stream>>>(x, wn1, h16);

  // 3) QKV GEMM with scatter epilogue (also fills fp32 K/V caches in d_out)
  gemm_kernel<<<dim3(3 * kD / 128, kM / 128), 256, 0, stream>>>(
      h16, wqkv16, kM, 3 * kD, kD, EP_QKV, nullptr, nullptr, q16, k16, v16, kcache,
      vcache);

  // 4) causal flash attention
  attn_kernel<<<kB * kH * (kT / 64), 128, 0, stream>>>(q16, k16, v16, attn16);

  // 5) proj GEMM + residual -> x1
  gemm_kernel<<<dim3(kD / 128, kM / 128), 256, 0, stream>>>(
      attn16, wproj16, kM, kD, kD, EP_RESID, x, x1, nullptr, nullptr, nullptr, nullptr,
      nullptr);

  // 6) rmsnorm2 (reuse h buffer)
  rmsnorm_kernel<<<kM, 256, 0, stream>>>(x1, wn2, h16);

  // 7) fused gate/up GEMM + SwiGLU -> gu16
  gateup_kernel<<<dim3(4 * kD / 128, kM / 128), 256, 0, stream>>>(h16, wgate16, wup16,
                                                                  gu16, 4 * kD, kD);

  // 8) down GEMM + residual -> d_out x region
  gemm_kernel<<<dim3(kD / 128, kM / 128), 256, 0, stream>>>(
      gu16, wdown16, kM, kD, 4 * kD, EP_RESID, x1, out_x, nullptr, nullptr, nullptr,
      nullptr, nullptr);
}
